// RNN_custom_21646635172460
// MI455X (gfx1250) — matrix operand order repaired and checked by reading, not compiled, not run on hardware
//
#include <hip/hip_runtime.h>
#include <hip/hip_bf16.h>

// ---------------------------------------------------------------------------
// Adaptive-LIF spiking RNN for MI455X (gfx1250, wave32, WMMA).
//   - hoisted input projection: large bf16 WMMA GEMM (102400x256, K=704),
//     K-loop split into unguarded vectorized body + single guarded tail,
//     4 N-tiles per wave (A fragment reused 4x).
//   - per-timestep recurrent GEMMs: 1 wave == one 16x16 f32 tile,
//     v_wmma_f32_16x16x32_bf16 over K=256 (8 wmma) / K=512 (16 wmma).
//   - operand layouts: contiguous k = 16*half + s in A and B (A: two 16B loads/lane,
//     B: one 32B load/lane, W row-major [N,K], no transpose loads needed).
// ---------------------------------------------------------------------------

typedef __attribute__((ext_vector_type(16))) __bf16 v16bf;
typedef __attribute__((ext_vector_type(8)))  __bf16 v8bf;
typedef __attribute__((ext_vector_type(8)))  float  v8f;

#define BB    1024   // batch
#define TT    100    // timesteps
#define INDIM 700
#define INP   704    // padded K for input projection
#define KFULL 672    // last fully-in-bounds K block start is 640; loop bound
#define HH    256
#define OUTD  20
#define OUTP  32     // padded N for output layer

__device__ __forceinline__ v8f wmma_bf16(v16bf a, v16bf b, v8f c) {
  return __builtin_amdgcn_wmma_f32_16x16x32_bf16(false, a, false, b,
                                                 (short)0, c, false, false);
}

// S: spikes [M x K] row-major bf16;  W: weights [N x K] row-major bf16.
// Computes one 16x16 tile of S @ W^T accumulated into acc.
__device__ __forceinline__ v8f gemm_acc(const __bf16* __restrict__ S,
                                        const __bf16* __restrict__ W,
                                        int mtile, int ntile, int K, v8f acc) {
  const int lane = threadIdx.x & 31;
  const int half = lane >> 4;
  const int l    = lane & 15;
  const __bf16* srow = S + (size_t)(mtile * 16 + l) * K;
  const __bf16* wrow = W + (size_t)(ntile * 16 + l) * K;
  for (int kb = 0; kb < K; kb += 32) {
    // A 16x32: lane<16 holds K[kb..kb+16); lane>=16 holds K[kb+16..kb+32)
    v8bf a0 = *(const v8bf*)(srow + kb + half * 16);
    v8bf a1 = *(const v8bf*)(srow + kb + half * 16 + 8);
    v16bf a = __builtin_shufflevector(a0, a1, 0, 1, 2, 3, 4, 5, 6, 7,
                                      8, 9, 10, 11, 12, 13, 14, 15);
    // B 32x16: lane holds B(kb+half*16+e, n=l) = W[n,k] -> 16 contiguous bf16
    v16bf b = *(const v16bf*)(wrow + kb + half * 16);
    acc = wmma_bf16(a, b, acc);
  }
  return acc;
}

// mem_update_adp elementwise core (b_j0=0.01, beta=1.8)
__device__ __forceinline__ void neuron_update(float hin, float sp_prev,
                                              float alpha, float ro,
                                              float& mem, float& bst,
                                              float& spike_out) {
  float bv  = ro * bst + (1.0f - ro) * sp_prev;
  float Bth = 0.01f + 1.8f * bv;
  float m   = mem * alpha + (1.0f - alpha) * hin - Bth * sp_prev;
  spike_out = (m - Bth) > 0.0f ? 1.0f : 0.0f;
  mem = m;
  bst = bv;
}

// ---------------------------------------------------------------------------
// init: neuron states, exp(-1/tau) tables, zero sumsp (re-done every launch)
// ---------------------------------------------------------------------------
__global__ void __launch_bounds__(256)
k_init(const float* __restrict__ r1_init, const float* __restrict__ r2_init,
       const float* __restrict__ d1_init, const float* __restrict__ o_mem_init,
       const float* __restrict__ ta_r1, const float* __restrict__ tm_r1,
       const float* __restrict__ ta_r2, const float* __restrict__ tm_r2,
       const float* __restrict__ ta_d1, const float* __restrict__ tm_d1,
       const float* __restrict__ ta_o,  const float* __restrict__ tm_o,
       float* r1m, float* b1, float* r2m, float* b2, float* d1m, float* b3,
       float* om, float* bo, float* osp, float* sumsp,
       __bf16* r1s_bf1, __bf16* r2s_bf1, __bf16* d1s_bf,
       float* a1, float* ro1, float* a2, float* ro2, float* a3, float* ro3,
       float* ao, float* roo) {
  int i = blockIdx.x * blockDim.x + threadIdx.x;
  int b = i >> 8, h = i & 255;
  r1m[i] = r1_init[i]; b1[i] = 0.01f; r1s_bf1[i] = (__bf16)r1_init[i];
  r2m[i] = r2_init[i]; b2[i] = 0.01f; r2s_bf1[i] = (__bf16)r2_init[i];
  d1m[i] = d1_init[i]; b3[i] = 0.01f; d1s_bf[i]  = (__bf16)d1_init[i];
  if (h < OUTD) {
    int j = b * OUTD + h;
    om[j] = o_mem_init[j]; bo[j] = 0.01f; osp[j] = 0.0f; sumsp[j] = 0.0f;
  }
  if (b == 0) {
    a1[h] = __expf(-1.0f / tm_r1[h]); ro1[h] = __expf(-1.0f / ta_r1[h]);
    a2[h] = __expf(-1.0f / tm_r2[h]); ro2[h] = __expf(-1.0f / ta_r2[h]);
    a3[h] = __expf(-1.0f / tm_d1[h]); ro3[h] = __expf(-1.0f / ta_d1[h]);
    if (h < OUTD) {
      ao[h] = __expf(-1.0f / tm_o[h]); roo[h] = __expf(-1.0f / ta_o[h]);
    }
  }
}

// ---------------------------------------------------------------------------
// convert weights f32 -> bf16 (i2h padded to K=704, d2o padded to N=32)
// ---------------------------------------------------------------------------
__global__ void __launch_bounds__(256)
k_convw(const float* __restrict__ w_i2h, const float* __restrict__ w_h2h,
        const float* __restrict__ w_h2d, const float* __restrict__ w_d2d,
        const float* __restrict__ w_dense1, const float* __restrict__ w_d2o,
        __bf16* wi, __bf16* whh, __bf16* whd, __bf16* wdd, __bf16* wdn,
        __bf16* wdo) {
  int i = blockIdx.x * blockDim.x + threadIdx.x;
  if (i < HH * HH) {
    whh[i] = (__bf16)w_h2h[i];
    whd[i] = (__bf16)w_h2d[i];
    wdd[i] = (__bf16)w_d2d[i];
    wdn[i] = (__bf16)w_dense1[i];
  }
  if (i < HH * INP) {
    int r = i / INP, k = i % INP;
    wi[i] = (k < INDIM) ? (__bf16)w_i2h[r * INDIM + k] : (__bf16)0.0f;
  }
  if (i < OUTP * HH) {
    int n = i >> 8, k = i & 255;
    wdo[i] = (n < OUTD) ? (__bf16)w_d2o[n * HH + k] : (__bf16)0.0f;
  }
}

// ---------------------------------------------------------------------------
// hoisted input projection: Xproj[t,b,h] = x[b,t,:] @ w_i2h[h,:] + bias
// x rows contiguous in (b,t) order: row r -> x + r*700 (2800B = 16B aligned).
// One block == one M-tile; each of 4 waves owns 4 N-tiles (A reused 4x).
// Hot loop: unguarded float4 loads + cvt; single guarded tail block.
// ---------------------------------------------------------------------------
__global__ void __launch_bounds__(128)
k_proj(const float* __restrict__ x, const __bf16* __restrict__ wi,
       const float* __restrict__ b_i2h, const float* __restrict__ b_h2h,
       float* __restrict__ xproj) {
  const int mtile = blockIdx.x;           // 0..6399
  const int wv    = threadIdx.x >> 5;     // 0..3 -> N-tiles 4*wv..4*wv+3
  const int lane  = threadIdx.x & 31;
  const int half  = lane >> 4;
  const int l     = lane & 15;
  const float*  xrow = x + (size_t)(mtile * 16 + l) * INDIM;
  const __bf16* wr0 = wi + (size_t)((wv * 4 + 0) * 16 + l) * INP;
  const __bf16* wr1 = wi + (size_t)((wv * 4 + 1) * 16 + l) * INP;
  const __bf16* wr2 = wi + (size_t)((wv * 4 + 2) * 16 + l) * INP;
  const __bf16* wr3 = wi + (size_t)((wv * 4 + 3) * 16 + l) * INP;
  v8f acc0 = {}, acc1 = {}, acc2 = {}, acc3 = {};

  // 21 fully-in-bounds K blocks: max index = 640+16+8+7 = 671 < 700
  for (int kb = 0; kb < KFULL; kb += 32) {
    const float4* p0 = (const float4*)(xrow + kb + half * 16);
    const float4* p1 = (const float4*)(xrow + kb + half * 16 + 8);
    float4 f0 = p0[0], f1 = p0[1], f2 = p1[0], f3 = p1[1];
    v16bf a;
    a[0] = (__bf16)f0.x; a[1] = (__bf16)f0.y; a[2] = (__bf16)f0.z; a[3] = (__bf16)f0.w;
    a[4] = (__bf16)f1.x; a[5] = (__bf16)f1.y; a[6] = (__bf16)f1.z; a[7] = (__bf16)f1.w;
    a[8]  = (__bf16)f2.x; a[9]  = (__bf16)f2.y; a[10] = (__bf16)f2.z; a[11] = (__bf16)f2.w;
    a[12] = (__bf16)f3.x; a[13] = (__bf16)f3.y; a[14] = (__bf16)f3.z; a[15] = (__bf16)f3.w;
    v16bf b0 = *(const v16bf*)(wr0 + kb + half * 16);
    v16bf b1 = *(const v16bf*)(wr1 + kb + half * 16);
    v16bf b2 = *(const v16bf*)(wr2 + kb + half * 16);
    v16bf b3 = *(const v16bf*)(wr3 + kb + half * 16);
    acc0 = wmma_bf16(a, b0, acc0);
    acc1 = wmma_bf16(a, b1, acc1);
    acc2 = wmma_bf16(a, b2, acc2);
    acc3 = wmma_bf16(a, b3, acc3);
  }

  // tail block kb=672: group0 max k = 695 (safe); group1 needs guard (<=703)
  {
    const int kb = KFULL;
    v16bf a;
#pragma unroll
    for (int i = 0; i < 8; ++i) {
      int k0 = kb + half * 16 + i;         // 672..695, always valid
      int k1 = kb + half * 16 + 8 + i;     // 680..703, partially valid
      a[i]     = (__bf16)xrow[k0];
      a[8 + i] = (k1 < INDIM) ? (__bf16)xrow[k1] : (__bf16)0.0f;
    }
    v16bf b0 = *(const v16bf*)(wr0 + kb + half * 16);
    v16bf b1 = *(const v16bf*)(wr1 + kb + half * 16);
    v16bf b2 = *(const v16bf*)(wr2 + kb + half * 16);
    v16bf b3 = *(const v16bf*)(wr3 + kb + half * 16);
    acc0 = wmma_bf16(a, b0, acc0);
    acc1 = wmma_bf16(a, b1, acc1);
    acc2 = wmma_bf16(a, b2, acc2);
    acc3 = wmma_bf16(a, b3, acc3);
  }

  // epilogue: scatter 4 tiles; C layout: VGPR r -> M = half*8+r, N = l
#pragma unroll
  for (int j = 0; j < 4; ++j) {
    v8f acc = (j == 0) ? acc0 : (j == 1) ? acc1 : (j == 2) ? acc2 : acc3;
    const int n = (wv * 4 + j) * 16 + l;
    const float bias = b_i2h[n] + b_h2h[n];
#pragma unroll
    for (int r = 0; r < 8; ++r) {
      int row = mtile * 16 + half * 8 + r;  // (b,t) flat row
      int bb = row / TT, tt = row % TT;
      xproj[(size_t)tt * BB * HH + (size_t)bb * HH + n] = acc[r] + bias;
    }
  }
}

// ---------------------------------------------------------------------------
// layer 1: hin = Xproj[t] + r1s_prev @ w_h2h^T ; adaptive-LIF update
// 1024 tiles -> 256 blocks x 128 (max spread for the latency-bound chain)
// ---------------------------------------------------------------------------
__global__ void __launch_bounds__(128)
k_l1(const float* __restrict__ xproj_t, const __bf16* __restrict__ sp_prev,
     const __bf16* __restrict__ whh, const float* __restrict__ alpha,
     const float* __restrict__ ro, float* __restrict__ memb,
     float* __restrict__ bst, __bf16* __restrict__ sp_cur,
     float* __restrict__ out_slab) {
  const int wave  = blockIdx.x * 4 + (threadIdx.x >> 5);
  const int mtile = wave >> 4, ntile = wave & 15;
  const int lane  = threadIdx.x & 31, half = lane >> 4, l = lane & 15;
  v8f acc = {};
  acc = gemm_acc(sp_prev, whh, mtile, ntile, HH, acc);
  const int   n  = ntile * 16 + l;
  const float al = alpha[n], r_ = ro[n];
#pragma unroll
  for (int r = 0; r < 8; ++r) {
    int bb = mtile * 16 + half * 8 + r;
    size_t idx = (size_t)bb * HH + n;
    float hin = xproj_t[idx] + acc[r];
    float sp  = (float)sp_prev[idx];
    float m = memb[idx], bv = bst[idx], ns;
    neuron_update(hin, sp, al, r_, m, bv, ns);
    memb[idx] = m; bst[idx] = bv;
    sp_cur[idx] = (__bf16)ns;
    out_slab[idx] = ns;
  }
}

// ---------------------------------------------------------------------------
// layer 2: hin = r1s @ w_h2d^T + b_h2d + r2s_prev @ w_d2d^T + b_d2d
// ---------------------------------------------------------------------------
__global__ void __launch_bounds__(128)
k_l2(const __bf16* __restrict__ r1s, const __bf16* __restrict__ sp_prev,
     const __bf16* __restrict__ wa, const __bf16* __restrict__ wb,
     const float* __restrict__ bias_a, const float* __restrict__ bias_b,
     const float* __restrict__ alpha, const float* __restrict__ ro,
     float* __restrict__ memb, float* __restrict__ bst,
     __bf16* __restrict__ sp_cur, float* __restrict__ out_slab) {
  const int wave  = blockIdx.x * 4 + (threadIdx.x >> 5);
  const int mtile = wave >> 4, ntile = wave & 15;
  const int lane  = threadIdx.x & 31, half = lane >> 4, l = lane & 15;
  v8f acc = {};
  acc = gemm_acc(r1s, wa, mtile, ntile, HH, acc);
  acc = gemm_acc(sp_prev, wb, mtile, ntile, HH, acc);
  const int   n    = ntile * 16 + l;
  const float al   = alpha[n], r_ = ro[n];
  const float bias = bias_a[n] + bias_b[n];
#pragma unroll
  for (int r = 0; r < 8; ++r) {
    int bb = mtile * 16 + half * 8 + r;
    size_t idx = (size_t)bb * HH + n;
    float hin = bias + acc[r];
    float sp  = (float)sp_prev[idx];
    float m = memb[idx], bv = bst[idx], ns;
    neuron_update(hin, sp, al, r_, m, bv, ns);
    memb[idx] = m; bst[idx] = bv;
    sp_cur[idx] = (__bf16)ns;
    out_slab[idx] = ns;
  }
}

// ---------------------------------------------------------------------------
// layer 3 (dense1): hin = r2s @ w_dense1^T + b_dense1 ; spike buffer in-place
// ---------------------------------------------------------------------------
__global__ void __launch_bounds__(128)
k_l3(const __bf16* __restrict__ r2s, const __bf16* __restrict__ w,
     const float* __restrict__ bias, const float* __restrict__ alpha,
     const float* __restrict__ ro, float* __restrict__ memb,
     float* __restrict__ bst, __bf16* __restrict__ sp_io,
     float* __restrict__ out_slab) {
  const int wave  = blockIdx.x * 4 + (threadIdx.x >> 5);
  const int mtile = wave >> 4, ntile = wave & 15;
  const int lane  = threadIdx.x & 31, half = lane >> 4, l = lane & 15;
  v8f acc = {};
  acc = gemm_acc(r2s, w, mtile, ntile, HH, acc);
  const int   n  = ntile * 16 + l;
  const float al = alpha[n], r_ = ro[n], bia = bias[n];
#pragma unroll
  for (int r = 0; r < 8; ++r) {
    int bb = mtile * 16 + half * 8 + r;
    size_t idx = (size_t)bb * HH + n;
    float hin = bia + acc[r];
    float sp  = (float)sp_io[idx];  // previous d1 spike (same element, safe RMW)
    float m = memb[idx], bv = bst[idx], ns;
    neuron_update(hin, sp, al, r_, m, bv, ns);
    memb[idx] = m; bst[idx] = bv;
    sp_io[idx] = (__bf16)ns;
    out_slab[idx] = ns;
  }
}

// ---------------------------------------------------------------------------
// output layer: hin = d1s @ w_d2o^T + b_d2o (N padded to 32); sumsp accum t>0
// 128 waves -> 32 blocks x 128
// ---------------------------------------------------------------------------
__global__ void __launch_bounds__(128)
k_lo(const __bf16* __restrict__ d1s, const __bf16* __restrict__ wdo,
     const float* __restrict__ bias, const float* __restrict__ alpha,
     const float* __restrict__ ro, float* __restrict__ memb,
     float* __restrict__ bst, float* __restrict__ osp,
     float* __restrict__ out_slab, float* __restrict__ sumsp, int t) {
  const int wave  = blockIdx.x * 4 + (threadIdx.x >> 5);
  const int mtile = wave >> 1, ntile = wave & 1;
  const int lane  = threadIdx.x & 31, half = lane >> 4, l = lane & 15;
  v8f acc = {};
  acc = gemm_acc(d1s, wdo, mtile, ntile, HH, acc);  // EXEC all-1s here
  const int n = ntile * 16 + l;
  if (n < OUTD) {
    const float al = alpha[n], r_ = ro[n], bia = bias[n];
#pragma unroll
    for (int r = 0; r < 8; ++r) {
      int bb = mtile * 16 + half * 8 + r;
      size_t idx = (size_t)bb * OUTD + n;
      float hin = bia + acc[r];
      float sp  = osp[idx];
      float m = memb[idx], bv = bst[idx], ns;
      neuron_update(hin, sp, al, r_, m, bv, ns);
      memb[idx] = m; bst[idx] = bv; osp[idx] = ns;
      out_slab[idx] = ns;
      if (t > 0) sumsp[idx] += ns;
    }
  }
}

// ---------------------------------------------------------------------------
extern "C" void kernel_launch(void* const* d_in, const int* in_sizes, int n_in,
                              void* d_out, int out_size, void* d_ws,
                              size_t ws_size, hipStream_t stream) {
  (void)in_sizes; (void)n_in; (void)out_size; (void)ws_size;
  const float* x        = (const float*)d_in[0];
  const float* w_i2h    = (const float*)d_in[1];
  const float* b_i2h    = (const float*)d_in[2];
  const float* w_h2h    = (const float*)d_in[3];
  const float* b_h2h    = (const float*)d_in[4];
  const float* w_h2d    = (const float*)d_in[5];
  const float* b_h2d    = (const float*)d_in[6];
  const float* w_d2d    = (const float*)d_in[7];
  const float* b_d2d    = (const float*)d_in[8];
  const float* w_dense1 = (const float*)d_in[9];
  const float* b_dense1 = (const float*)d_in[10];
  const float* w_d2o    = (const float*)d_in[11];
  const float* b_d2o    = (const float*)d_in[12];
  const float* ta_r1    = (const float*)d_in[13];
  const float* tm_r1    = (const float*)d_in[14];
  const float* ta_r2    = (const float*)d_in[15];
  const float* tm_r2    = (const float*)d_in[16];
  const float* ta_d1    = (const float*)d_in[17];
  const float* tm_d1    = (const float*)d_in[18];
  const float* ta_o     = (const float*)d_in[19];
  const float* tm_o     = (const float*)d_in[20];
  const float* r1_init  = (const float*)d_in[21];
  const float* r2_init  = (const float*)d_in[22];
  const float* d1_init  = (const float*)d_in[23];
  const float* om_init  = (const float*)d_in[24];

  // d_out: sumsp[1024,20] | l1[100,1024,256] | l2 | l3 | lo[100,1024,20]
  float* out   = (float*)d_out;
  float* sumsp = out;
  float* l1    = sumsp + (size_t)BB * OUTD;
  float* l2    = l1 + (size_t)TT * BB * HH;
  float* l3    = l2 + (size_t)TT * BB * HH;
  float* lo    = l3 + (size_t)TT * BB * HH;

  // workspace carving (256B aligned slices)
  char*  ws  = (char*)d_ws;
  size_t off = 0;
  auto carve = [&](size_t bytes) -> void* {
    off = (off + 255) & ~(size_t)255;
    void* p = ws + off;
    off += bytes;
    return p;
  };
  float* xproj = (float*)carve((size_t)TT * BB * HH * 4);  // ~100 MB
  float* r1m = (float*)carve((size_t)BB * HH * 4);
  float* b1  = (float*)carve((size_t)BB * HH * 4);
  float* r2m = (float*)carve((size_t)BB * HH * 4);
  float* b2  = (float*)carve((size_t)BB * HH * 4);
  float* d1m = (float*)carve((size_t)BB * HH * 4);
  float* b3  = (float*)carve((size_t)BB * HH * 4);
  float* om  = (float*)carve((size_t)BB * OUTD * 4);
  float* bo  = (float*)carve((size_t)BB * OUTD * 4);
  float* osp = (float*)carve((size_t)BB * OUTD * 4);
  float* a1  = (float*)carve(HH * 4);
  float* ro1 = (float*)carve(HH * 4);
  float* a2  = (float*)carve(HH * 4);
  float* ro2 = (float*)carve(HH * 4);
  float* a3  = (float*)carve(HH * 4);
  float* ro3 = (float*)carve(HH * 4);
  float* ao  = (float*)carve(OUTD * 4);
  float* roo = (float*)carve(OUTD * 4);
  __bf16* wi_bf  = (__bf16*)carve((size_t)HH * INP * 2);
  __bf16* whh_bf = (__bf16*)carve((size_t)HH * HH * 2);
  __bf16* whd_bf = (__bf16*)carve((size_t)HH * HH * 2);
  __bf16* wdd_bf = (__bf16*)carve((size_t)HH * HH * 2);
  __bf16* wdn_bf = (__bf16*)carve((size_t)HH * HH * 2);
  __bf16* wdo_bf = (__bf16*)carve((size_t)OUTP * HH * 2);
  __bf16* r1s_bf = (__bf16*)carve((size_t)2 * BB * HH * 2);  // ping-pong
  __bf16* r2s_bf = (__bf16*)carve((size_t)2 * BB * HH * 2);  // ping-pong
  __bf16* d1s_bf = (__bf16*)carve((size_t)BB * HH * 2);

  // 1) init states (t=0 reads ping-pong slot 1) + tau tables + sumsp=0
  k_init<<<(BB * HH) / 256, 256, 0, stream>>>(
      r1_init, r2_init, d1_init, om_init,
      ta_r1, tm_r1, ta_r2, tm_r2, ta_d1, tm_d1, ta_o, tm_o,
      r1m, b1, r2m, b2, d1m, b3, om, bo, osp, sumsp,
      r1s_bf + (size_t)BB * HH, r2s_bf + (size_t)BB * HH, d1s_bf,
      a1, ro1, a2, ro2, a3, ro3, ao, roo);

  // 2) weights -> bf16 (with padding)
  k_convw<<<(HH * INP) / 256, 256, 0, stream>>>(
      w_i2h, w_h2h, w_h2d, w_d2d, w_dense1, w_d2o,
      wi_bf, whh_bf, whd_bf, wdd_bf, wdn_bf, wdo_bf);

  // 3) hoisted input projection: one block per M-tile, 4 N-tiles per wave
  k_proj<<<(BB * TT) / 16, 128, 0, stream>>>(x, wi_bf, b_i2h, b_h2h, xproj);

  // 4) sequential timestep chain (kernel boundaries = grid-wide ordering)
  for (int t = 0; t < TT; ++t) {
    const __bf16* r1prev = r1s_bf + (size_t)((t + 1) & 1) * BB * HH;
    __bf16*       r1cur  = r1s_bf + (size_t)(t & 1) * BB * HH;
    const __bf16* r2prev = r2s_bf + (size_t)((t + 1) & 1) * BB * HH;
    __bf16*       r2cur  = r2s_bf + (size_t)(t & 1) * BB * HH;

    k_l1<<<256, 128, 0, stream>>>(xproj + (size_t)t * BB * HH, r1prev, whh_bf,
                                  a1, ro1, r1m, b1, r1cur,
                                  l1 + (size_t)t * BB * HH);
    k_l2<<<256, 128, 0, stream>>>(r1cur, r2prev, whd_bf, wdd_bf, b_h2d, b_d2d,
                                  a2, ro2, r2m, b2, r2cur,
                                  l2 + (size_t)t * BB * HH);
    k_l3<<<256, 128, 0, stream>>>(r2cur, wdn_bf, b_dense1, a3, ro3, d1m, b3,
                                  d1s_bf, l3 + (size_t)t * BB * HH);
    k_lo<<<32, 128, 0, stream>>>(d1s_bf, wdo_bf, b_d2o, ao, roo, om, bo, osp,
                                 lo + (size_t)t * BB * OUTD, sumsp, t);
  }
}
